// MyVMLMFCell_67027259621386
// MI455X (gfx1250) — matrix operand order repaired and checked by reading, not compiled, not run on hardware
//
#include <hip/hip_runtime.h>
#include <hip/hip_bf16.h>
#include <cstdint>
#include <cstddef>

// Problem constants (match reference)
static constexpr int kB = 8192;   // batch
static constexpr int kI = 512;    // input size
static constexpr int kH = 1024;   // hidden size
static constexpr int kR = 64;     // rank
static constexpr int kK = 128;    // concatenated rank dim (2*kR)

typedef __attribute__((ext_vector_type(16))) __bf16 v16bf;
typedef __attribute__((ext_vector_type(8)))  __bf16 v8bf;
typedef __attribute__((ext_vector_type(8)))  float  v8f;

__device__ __forceinline__ v16bf pack16(v8bf lo, v8bf hi) {
  v16bf r;
#pragma unroll
  for (int i = 0; i < 8; ++i) { r[i] = lo[i]; r[8 + i] = hi[i]; }
  return r;
}

// ---------------------------------------------------------------------------
// Prep 1: build bf16 transposed/concatenated weight buffers.
//   uxT[n][i] = u_x[i][n]        (64 x 512)
//   uhT[n][j] = u_h[j][n]        (64 x 1024)
//   W[n][0:64]  = v_x[n][:], W[n][64:128] = v_h[n][:]   (4096 x 128)
// ---------------------------------------------------------------------------
__global__ void prep_pack(const float* __restrict__ u_x, const float* __restrict__ u_h,
                          const float* __restrict__ v_x, const float* __restrict__ v_h,
                          __bf16* __restrict__ uxT, __bf16* __restrict__ uhT,
                          __bf16* __restrict__ W) {
  int t = blockIdx.x * blockDim.x + threadIdx.x;
  if (t < kR * kI) {
    int n = t / kI, i = t % kI;
    uxT[n * kI + i] = (__bf16)u_x[i * kR + n];
    return;
  }
  t -= kR * kI;
  if (t < kR * kH) {
    int n = t / kH, j = t % kH;
    uhT[n * kH + j] = (__bf16)u_h[j * kR + n];
    return;
  }
  t -= kR * kH;
  if (t < 4 * kH * kR) {
    int n = t / kR, r = t % kR;
    W[n * kK + r]      = (__bf16)v_x[n * kR + r];
    W[n * kK + kR + r] = (__bf16)v_h[n * kR + r];
  }
}

// ---------------------------------------------------------------------------
// Prep 2: rank-dim dot products for the diagonal-correction ("ref") terms.
//   coef_x[g][i] = sum_r u_x[i][r] * v_x[g*H+i][r]
//   coef_h[g][j] = sum_r u_h[j][r] * v_h[g*H+j][r]
// ---------------------------------------------------------------------------
__global__ void prep_coef(const float* __restrict__ u_x, const float* __restrict__ u_h,
                          const float* __restrict__ v_x, const float* __restrict__ v_h,
                          float* __restrict__ coef_x, float* __restrict__ coef_h) {
  int t = blockIdx.x * blockDim.x + threadIdx.x;
  if (t < 4 * kI) {
    int g = t / kI, i = t % kI;
    float s = 0.f;
#pragma unroll 8
    for (int r = 0; r < kR; ++r) s += u_x[i * kR + r] * v_x[(g * kH + i) * kR + r];
    coef_x[t] = s;
  } else if (t < 4 * kI + 4 * kH) {
    int u = t - 4 * kI;
    int g = u / kH, j = u % kH;
    float s = 0.f;
#pragma unroll 8
    for (int r = 0; r < kR; ++r) s += u_h[j * kR + r] * v_h[(g * kH + j) * kR + r];
    coef_h[u] = s;
  }
}

// ---------------------------------------------------------------------------
// Stage 1: P[b][0:64] = x @ u_x, P[b][64:128] = h @ u_h  (bf16 out, f32 acc).
// One wave computes a 16-row x 64-col strip; K-loop in steps of 32 with
// v_wmma_f32_16x16x32_bf16. A operand converted f32->bf16 on the fly.
// ---------------------------------------------------------------------------
__global__ __launch_bounds__(256) void lowrank_proj(
    const float* __restrict__ x, const float* __restrict__ h,
    const __bf16* __restrict__ uxT, const __bf16* __restrict__ uhT,
    __bf16* __restrict__ P) {
  const int wid  = blockIdx.x * (blockDim.x >> 5) + (threadIdx.x >> 5);
  const int lane = threadIdx.x & 31;
  const int half = lane >> 4;
  const int mrow = lane & 15;     // A row / B col / D col, per ISA layout

  const float* src; const __bf16* wT; int K, noff, rowbase;
  if (wid < kB / 16) { src = x; wT = uxT; K = kI; noff = 0;  rowbase = wid * 16; }
  else               { src = h; wT = uhT; K = kH; noff = kR; rowbase = (wid - kB / 16) * 16; }

  v8f acc[4] = {};
  const float* arow = src + (size_t)(rowbase + mrow) * K;

  for (int kb = 0; kb < K; kb += 32) {
    // A: lane holds K = {kb+half*8 .. +7} and {kb+16+half*8 .. +7} (two f32x8 chunks)
    const float4* p0 = (const float4*)(arow + kb + half * 8);
    const float4* p1 = (const float4*)(arow + kb + 16 + half * 8);
    float4 f0 = p0[0], f1 = p0[1], f2 = p1[0], f3 = p1[1];
    v8bf alo, ahi;
    alo[0]=(__bf16)f0.x; alo[1]=(__bf16)f0.y; alo[2]=(__bf16)f0.z; alo[3]=(__bf16)f0.w;
    alo[4]=(__bf16)f1.x; alo[5]=(__bf16)f1.y; alo[6]=(__bf16)f1.z; alo[7]=(__bf16)f1.w;
    ahi[0]=(__bf16)f2.x; ahi[1]=(__bf16)f2.y; ahi[2]=(__bf16)f2.z; ahi[3]=(__bf16)f2.w;
    ahi[4]=(__bf16)f3.x; ahi[5]=(__bf16)f3.y; ahi[6]=(__bf16)f3.z; ahi[7]=(__bf16)f3.w;
    v16bf a = pack16(alo, ahi);
#pragma unroll
    for (int nt = 0; nt < 4; ++nt) {
      // B: lane holds col N = mrow, K = {kb+half*8 .. +7} and {kb+16+half*8 .. +7}
      const __bf16* brow = wT + (size_t)(nt * 16 + mrow) * K + kb + half * 8;
      v16bf b = pack16(*(const v8bf*)brow, *(const v8bf*)(brow + 16));
      acc[nt] = __builtin_amdgcn_wmma_f32_16x16x32_bf16(
          false, a, false, b, (short)0, acc[nt], false, false);
    }
  }
  // D layout: lane has col N = mrow, rows M = half*8 + r
#pragma unroll
  for (int nt = 0; nt < 4; ++nt)
#pragma unroll
    for (int r = 0; r < 8; ++r)
      P[(size_t)(rowbase + half * 8 + r) * kK + noff + nt * 16 + mrow] = (__bf16)acc[nt][r];
}

// ---------------------------------------------------------------------------
// Stage 2 (fused): lowered = P @ W^T for all 4 gates of one 16x16 tile,
// then in-register LSTM gate epilogue -> h_next, c_next. Never materializes
// the [B,4H] intermediates (saves ~256 MB of HBM round trips).
// ---------------------------------------------------------------------------
__global__ __launch_bounds__(256) void fused_gates(
    const __bf16* __restrict__ P, const __bf16* __restrict__ W,
    const float* __restrict__ x, const float* __restrict__ h, const float* __restrict__ c,
    const float* __restrict__ coef_x, const float* __restrict__ coef_h,
    const float* __restrict__ b_x, const float* __restrict__ b_h,
    const float* __restrict__ dia_x, const float* __restrict__ dia_h,
    float* __restrict__ out) {
  const int wid  = blockIdx.x * (blockDim.x >> 5) + (threadIdx.x >> 5);
  const int lane = threadIdx.x & 31;
  const int half = lane >> 4;
  const int mrow = lane & 15;
  const int rt = wid >> 6;             // batch tile (0..511)
  const int ct = wid & 63;             // hidden-col tile (0..63); consecutive waves share P rows
  const int rowbase = rt * 16;
  const int colbase = ct * 16;

  v8f acc[4] = {};
  const __bf16* arow = P + (size_t)(rowbase + mrow) * kK;
#pragma unroll
  for (int kb = 0; kb < kK; kb += 32) {
    v16bf a = pack16(*(const v8bf*)(arow + kb + half * 8),
                     *(const v8bf*)(arow + kb + 16 + half * 8));
#pragma unroll
    for (int g = 0; g < 4; ++g) {
      const __bf16* brow = W + (size_t)(g * kH + colbase + mrow) * kK + kb + half * 8;
      v16bf b = pack16(*(const v8bf*)brow, *(const v8bf*)(brow + 16));
      acc[g] = __builtin_amdgcn_wmma_f32_16x16x32_bf16(
          false, a, false, b, (short)0, acc[g], false, false);
    }
  }

  // Epilogue. D layout: this lane owns column `col`, batch rows half*8 + r.
  const int col = colbase + mrow;
  const bool inI = (col < kI);
  const float dx = inI ? dia_x[col] : 0.f;
  const float dh = dia_h[col];
  float cfx[4], cfh[4], bb[4];
#pragma unroll
  for (int g = 0; g < 4; ++g) {
    cfx[g] = inI ? coef_x[g * kI + col] : 0.f;
    cfh[g] = coef_h[g * kH + col];
    bb[g]  = b_x[g * kH + col] + b_h[g * kH + col];
  }
#pragma unroll
  for (int r = 0; r < 8; ++r) {
    const int b = rowbase + half * 8 + r;
    const float xv = inI ? x[(size_t)b * kI + col] : 0.f;
    const float hv = h[(size_t)b * kH + col];
    const float cv = c[(size_t)b * kH + col];
    const float z  = dx * xv + dh * hv;
    float pre[4];
#pragma unroll
    for (int g = 0; g < 4; ++g)
      pre[g] = acc[g][r] - xv * cfx[g] - hv * cfh[g] + bb[g] + z;
    const float ig = 1.f / (1.f + __expf(-pre[0]));
    const float fg = 1.f / (1.f + __expf(-pre[1]));
    const float og = 1.f / (1.f + __expf(-pre[2]));
    const float ng = tanhf(pre[3]);
    const float cn = fg * cv + ig * ng;
    const float hn = og * tanhf(cn);
    out[(size_t)b * kH + col] = hn;                           // h_next
    out[(size_t)kB * kH + (size_t)b * kH + col] = cn;         // c_next
  }
}

// ---------------------------------------------------------------------------
extern "C" void kernel_launch(void* const* d_in, const int* in_sizes, int n_in,
                              void* d_out, int out_size, void* d_ws, size_t ws_size,
                              hipStream_t stream) {
  const float* x     = (const float*)d_in[0];
  const float* h     = (const float*)d_in[1];
  const float* c     = (const float*)d_in[2];
  const float* u_x   = (const float*)d_in[3];
  const float* u_h   = (const float*)d_in[4];
  const float* v_x   = (const float*)d_in[5];
  const float* v_h   = (const float*)d_in[6];
  const float* b_x   = (const float*)d_in[7];
  const float* b_h   = (const float*)d_in[8];
  const float* dia_x = (const float*)d_in[9];
  const float* dia_h = (const float*)d_in[10];
  float* out = (float*)d_out;

  // Workspace layout (all offsets 256B-aligned; ~3.4 MB total)
  char* ws = (char*)d_ws;
  __bf16* uxT   = (__bf16*)(ws);               //  64*512*2  =  64 KB
  __bf16* uhT   = (__bf16*)(ws + 65536);       //  64*1024*2 = 128 KB
  __bf16* W     = (__bf16*)(ws + 196608);      // 4096*128*2 =   1 MB
  float*  coefx = (float*) (ws + 1245184);     // 4*512*4    =   8 KB
  float*  coefh = (float*) (ws + 1253376);     // 4*1024*4   =  16 KB
  __bf16* P     = (__bf16*)(ws + 1269760);     // 8192*128*2 =   2 MB

  {
    int total = kR * kI + kR * kH + 4 * kH * kR;
    prep_pack<<<(total + 255) / 256, 256, 0, stream>>>(u_x, u_h, v_x, v_h, uxT, uhT, W);
  }
  {
    int total = 4 * kI + 4 * kH;
    prep_coef<<<(total + 255) / 256, 256, 0, stream>>>(u_x, u_h, v_x, v_h, coefx, coefh);
  }
  {
    int waves = 2 * (kB / 16);                 // 1024 waves, 8 waves/block
    lowrank_proj<<<waves / 8, 256, 0, stream>>>(x, h, uxT, uhT, P);
  }
  {
    int waves = (kB / 16) * (kH / 16);         // 32768 waves, 8 waves/block
    fused_gates<<<waves / 8, 256, 0, stream>>>(P, W, x, h, c, coefx, coefh,
                                               b_x, b_h, dia_x, dia_h, out);
  }
}
